// DeeperGCN_52364241273197
// MI455X (gfx1250) — matrix operand order repaired and checked by reading, not compiled, not run on hardware
//
#include <hip/hip_runtime.h>
#include <hip/hip_bf16.h>
#include <stdint.h>

#define D_F 128
#define D_H 256
#define GEPS 1e-7f

typedef unsigned int  u32;
typedef unsigned short u16;
typedef __attribute__((ext_vector_type(16))) __bf16 v16bf;
typedef __attribute__((ext_vector_type(8)))  float  v8f;
typedef __attribute__((ext_vector_type(8)))  u32    v8u;

// fp32 -> bf16 round-to-nearest-even
static __device__ __forceinline__ u16 f2bf(float f) {
  u32 u = __float_as_uint(f);
  u += 0x7FFFu + ((u >> 16) & 1u);
  return (u16)(u >> 16);
}

// ---------------------------------------------------------------------------
// Pack a row-major fp32 [K, Nn] weight matrix into WMMA bf16 B-fragment order,
// lane-major so each lane's 8 fragment dwords are 32 contiguous bytes:
//   P[(t*32 + lane)*8 + v] = pack(bf16(W[k0][n]), bf16(W[k0+1][n]))
// with t = kt*NT + nt, n = nt*16 + (lane&15), k0 = kt*32 + (lane>>4)*16 + 2v
// (contiguous k order, same as A: lanes 0-15 hold K 0-15 pairs, lanes 16-31 K 16-31)
// ---------------------------------------------------------------------------
__global__ void pack_b_kernel(const float* __restrict__ W, u32* __restrict__ P,
                              int K, int Nn) {
  const int NT = Nn >> 4;
  const int total = (K >> 5) * NT * 256;
  for (int i = blockIdx.x * blockDim.x + threadIdx.x; i < total;
       i += gridDim.x * blockDim.x) {
    int v    = i & 7;
    int lane = (i >> 3) & 31;
    int t    = i >> 8;
    int nt   = t % NT, kt = t / NT;
    int n    = nt * 16 + (lane & 15);
    int k0   = kt * 32 + (lane >> 4) * 16 + v * 2;
    u32 lo = f2bf(W[(size_t)k0 * Nn + n]);
    u32 hi = f2bf(W[(size_t)(k0 + 1) * Nn + n]);
    P[i] = lo | (hi << 16);
  }
}

// ---------------------------------------------------------------------------
// Edge pass 1: per-dst per-feature max of msg = relu(x[src])+eps.
// msg > 0 always, so int atomicMax on the float bit pattern is exact and the
// 0-initialized buffer reproduces the reference's isolated-node guard (m=0).
// ---------------------------------------------------------------------------
__global__ void edge_max_kernel(const float* __restrict__ x,
                                const int* __restrict__ src,
                                const int* __restrict__ dst,
                                int* __restrict__ mI, int E) {
  int t = blockIdx.x * blockDim.x + threadIdx.x;
  if (t >= E * 32) return;
  int e = t >> 5;
  int c = (t & 31) * 4;
  int s = src[e], d = dst[e];
  float4 xv = *(const float4*)(x + (size_t)s * D_F + c);
  int* mp = mI + (size_t)d * D_F + c;
  atomicMax(mp + 0, __float_as_int(fmaxf(xv.x, 0.f) + GEPS));
  atomicMax(mp + 1, __float_as_int(fmaxf(xv.y, 0.f) + GEPS));
  atomicMax(mp + 2, __float_as_int(fmaxf(xv.z, 0.f) + GEPS));
  atomicMax(mp + 3, __float_as_int(fmaxf(xv.w, 0.f) + GEPS));
}

// ---------------------------------------------------------------------------
// Edge pass 2: e = exp(msg - m[dst]); accumulate sumE += e, sumME += msg*e.
// (alpha has a common per-dst denominator -> fold passes 2 and 3 of reference)
// ---------------------------------------------------------------------------
__global__ void edge_sum_kernel(const float* __restrict__ x,
                                const int* __restrict__ src,
                                const int* __restrict__ dst,
                                const float* __restrict__ mF,
                                float* __restrict__ sE,
                                float* __restrict__ sME, int E) {
  int t = blockIdx.x * blockDim.x + threadIdx.x;
  if (t >= E * 32) return;
  int e = t >> 5;
  int c = (t & 31) * 4;
  int s = src[e], d = dst[e];
  float4 xv = *(const float4*)(x  + (size_t)s * D_F + c);
  float4 mv = *(const float4*)(mF + (size_t)d * D_F + c);
  size_t o = (size_t)d * D_F + c;
  float msg, ev;
  msg = fmaxf(xv.x, 0.f) + GEPS; ev = __expf(msg - mv.x);
  atomicAdd(sE + o + 0, ev); atomicAdd(sME + o + 0, msg * ev);
  msg = fmaxf(xv.y, 0.f) + GEPS; ev = __expf(msg - mv.y);
  atomicAdd(sE + o + 1, ev); atomicAdd(sME + o + 1, msg * ev);
  msg = fmaxf(xv.z, 0.f) + GEPS; ev = __expf(msg - mv.z);
  atomicAdd(sE + o + 2, ev); atomicAdd(sME + o + 2, msg * ev);
  msg = fmaxf(xv.w, 0.f) + GEPS; ev = __expf(msg - mv.w);
  atomicAdd(sE + o + 3, ev); atomicAdd(sME + o + 3, msg * ev);
}

// ---------------------------------------------------------------------------
// Node epilogue: h = sumME/(sumE + 1e-16) + x, quantized to bf16 (WMMA A input)
// vectorized: float4 in, 4 packed bf16 (uint2) out
// ---------------------------------------------------------------------------
__global__ void node_finish_kernel(const float* __restrict__ x,
                                   const float* __restrict__ sE,
                                   const float* __restrict__ sME,
                                   u16* __restrict__ hbf, int total4) {
  int i = blockIdx.x * blockDim.x + threadIdx.x;
  if (i >= total4) return;
  size_t o = (size_t)i * 4;
  float4 a = *(const float4*)(sME + o);
  float4 s = *(const float4*)(sE + o);
  float4 xv = *(const float4*)(x + o);
  u32 h0 = f2bf(a.x / (s.x + 1e-16f) + xv.x);
  u32 h1 = f2bf(a.y / (s.y + 1e-16f) + xv.y);
  u32 h2 = f2bf(a.z / (s.z + 1e-16f) + xv.z);
  u32 h3 = f2bf(a.w / (s.w + 1e-16f) + xv.w);
  *(uint2*)(hbf + o) = make_uint2(h0 | (h1 << 16), h2 | (h3 << 16));
}

// ---------------------------------------------------------------------------
// Fused MLP: [32 rows] x  (A@W1 + b1 -> LayerNorm -> ReLU -> @W2 + b2)
// 256 threads = 8 wave32 waves; WMMA bf16 16x16x32, fp32 accumulation.
// ---------------------------------------------------------------------------
template <int DOUT, bool RELU_OUT>
__global__ __launch_bounds__(256) void mlp_kernel(
    const u16* __restrict__ Abf,   // [nrows,128] bf16 row-major
    const u32* __restrict__ W1p,   // packed B frags, K=128, N=256
    const float* __restrict__ b1,
    const float* __restrict__ gam, const float* __restrict__ bet,
    const u32* __restrict__ W2p,   // packed B frags, K=256, N=DOUT
    const float* __restrict__ b2,
    float* __restrict__ out, int nrows) {
  constexpr int AS = 136;  // bf16 stride (16B-aligned rows, bank-staggered)
  constexpr int HS = 257;  // f32 stride (bank-staggered)
  constexpr int BS = 258;  // bf16 stride (dword-aligned pairs, bank-staggered)
  __shared__ u16   A_lds[32 * AS];
  __shared__ float H[32 * HS];
  __shared__ u16   HB[32 * BS];

  const int tid  = threadIdx.x;
  const int lane = tid & 31;
  const int w    = tid >> 5;
  const int nlo  = lane & 15;   // n / m-in-halfwave index
  const int hi   = lane >> 4;
  const int blockRow = blockIdx.x * 32;

  // ---- stage A rows (32 x 128 bf16) into LDS, zero-fill the row tail ----
  {
    int c   = tid;              // 256 chunks of 16 bf16
    int row = c >> 3;
    int col = (c & 7) * 16;
    int grow = blockRow + row;
    uint4 d0 = make_uint4(0, 0, 0, 0), d1 = d0;
    if (grow < nrows) {
      const uint4* p = (const uint4*)(Abf + (size_t)grow * D_F + col);
      d0 = p[0];
      d1 = p[1];
    }
    uint4* q = (uint4*)(A_lds + row * AS + col);
    q[0] = d0;
    q[1] = d1;
  }
  __syncthreads();

  // ---- GEMM1: (32x128) @ (128x256) ----
  const int mt = w & 1;
  v8u a[4];
#pragma unroll
  for (int kt = 0; kt < 4; ++kt)
#pragma unroll
    for (int v = 0; v < 8; ++v) {
      int k = kt * 32 + hi * 16 + v * 2;  // contiguous k = 16*hi + s (matches packed B)
      a[kt][v] = *(const u32*)(A_lds + (mt * 16 + nlo) * AS + k);
    }

#pragma unroll
  for (int j = 0; j < 4; ++j) {
    int nt = (w >> 1) + 4 * j;
    v8f c = {0.f, 0.f, 0.f, 0.f, 0.f, 0.f, 0.f, 0.f};
#pragma unroll
    for (int kt = 0; kt < 4; ++kt) {
      // one 32-byte vector load per lane: 2x global_load_b128, fully coalesced
      v8u b = *(const v8u*)(W1p + (size_t)((kt * 16 + nt) * 32 + lane) * 8);
      c = __builtin_amdgcn_wmma_f32_16x16x32_bf16(
          false, __builtin_bit_cast(v16bf, a[kt]),
          false, __builtin_bit_cast(v16bf, b), (short)0, c, false, false);
    }
#pragma unroll
    for (int r = 0; r < 8; ++r) {  // D layout: m = r + hi*8, n = lane&15
      int mm = mt * 16 + hi * 8 + r;
      int nn = nt * 16 + nlo;
      H[mm * HS + nn] = c[r] + b1[nn];
    }
  }
  __syncthreads();

  // ---- LayerNorm + ReLU -> HB (bf16), 8 lanes per row ----
  {
    int row  = tid >> 3;
    int part = tid & 7;
    float s = 0.f, ss = 0.f;
#pragma unroll 8
    for (int ci = part * 32; ci < part * 32 + 32; ++ci) {
      float h = H[row * HS + ci];
      s += h;
      ss += h * h;
    }
    s += __shfl_xor(s, 1); ss += __shfl_xor(ss, 1);
    s += __shfl_xor(s, 2); ss += __shfl_xor(ss, 2);
    s += __shfl_xor(s, 4); ss += __shfl_xor(ss, 4);
    float mu   = s * (1.f / D_H);
    float var  = ss * (1.f / D_H) - mu * mu;
    float rstd = rsqrtf(var + 1e-5f);
    for (int ci = part * 32; ci < part * 32 + 32; ++ci) {
      float h = (H[row * HS + ci] - mu) * rstd * gam[ci] + bet[ci];
      HB[row * BS + ci] = f2bf(fmaxf(h, 0.f));
    }
  }
  __syncthreads();

  // ---- GEMM2: (32x256) @ (256xDOUT) ----
  constexpr int NT2 = DOUT / 16;
  constexpr int JT  = (NT2 + 3) / 4;
  v8f c2[JT];
#pragma unroll
  for (int j = 0; j < JT; ++j)
    c2[j] = (v8f){0.f, 0.f, 0.f, 0.f, 0.f, 0.f, 0.f, 0.f};
#pragma unroll
  for (int kt = 0; kt < 8; ++kt) {
    v8u a2;
#pragma unroll
    for (int v = 0; v < 8; ++v) {
      int k = kt * 32 + hi * 16 + v * 2;
      a2[v] = *(const u32*)(HB + (mt * 16 + nlo) * BS + k);
    }
#pragma unroll
    for (int j = 0; j < JT; ++j) {
      int nt = (w >> 1) + 4 * j;
      if (nt < NT2) {
        v8u b = *(const v8u*)(W2p + (size_t)((kt * NT2 + nt) * 32 + lane) * 8);
        c2[j] = __builtin_amdgcn_wmma_f32_16x16x32_bf16(
            false, __builtin_bit_cast(v16bf, a2),
            false, __builtin_bit_cast(v16bf, b), (short)0, c2[j], false, false);
      }
    }
  }
#pragma unroll
  for (int j = 0; j < JT; ++j) {
    int nt = (w >> 1) + 4 * j;
    if (nt < NT2) {
#pragma unroll
      for (int r = 0; r < 8; ++r) {
        int mm   = mt * 16 + hi * 8 + r;
        int grow = blockRow + mm;
        if (grow < nrows) {
          int nn = nt * 16 + nlo;
          float val = c2[j][r] + b2[nn];
          if (RELU_OUT) val = fmaxf(val, 0.f);
          out[(size_t)grow * DOUT + nn] = val;
        }
      }
    }
  }
}

// ---------------------------------------------------------------------------
extern "C" void kernel_launch(void* const* d_in, const int* in_sizes, int n_in,
                              void* d_out, int out_size, void* d_ws,
                              size_t ws_size, hipStream_t stream) {
  (void)n_in; (void)out_size; (void)ws_size;
  const float* x    = (const float*)d_in[0];
  const int*   ei   = (const int*)d_in[1];
  const float* W1_0 = (const float*)d_in[3];
  const float* b1_0 = (const float*)d_in[4];
  const float* g_0  = (const float*)d_in[5];
  const float* be_0 = (const float*)d_in[6];
  const float* W2_0 = (const float*)d_in[7];
  const float* b2_0 = (const float*)d_in[8];
  const float* W1_1 = (const float*)d_in[9];
  const float* b1_1 = (const float*)d_in[10];
  const float* g_1  = (const float*)d_in[11];
  const float* be_1 = (const float*)d_in[12];
  const float* W2_1 = (const float*)d_in[13];
  const float* b2_1 = (const float*)d_in[14];

  const int N = in_sizes[0] / D_F;
  const int E = in_sizes[2];
  const int* src  = ei;
  const int* dstp = ei + E;
  const size_t NF = (size_t)N * D_F;

  // workspace layout (all 16B-aligned)
  char* w = (char*)d_ws;
  int*   mI  = (int*)w;  float* mF = (float*)w;  w += NF * 4;  // segment max
  float* sE  = (float*)w;                        w += NF * 4;  // sum exp
  float* sME = (float*)w;                        w += NF * 4;  // sum msg*exp
  u16*   hbf = (u16*)w;                          w += NF * 2;  // MLP input bf16
  float* x1  = (float*)w;                        w += NF * 4;  // relu(layer0 out)
  u32* W10p = (u32*)w; w += 16384 * 4;
  u32* W20p = (u32*)w; w += 16384 * 4;
  u32* W11p = (u32*)w; w += 16384 * 4;
  u32* W21p = (u32*)w; w += 8192 * 4;

  // pack weights into WMMA B-fragment layout (bf16 pairs, lane-major)
  pack_b_kernel<<<64, 256, 0, stream>>>(W1_0, W10p, 128, 256);
  pack_b_kernel<<<64, 256, 0, stream>>>(W2_0, W20p, 256, 128);
  pack_b_kernel<<<64, 256, 0, stream>>>(W1_1, W11p, 128, 256);
  pack_b_kernel<<<32, 256, 0, stream>>>(W2_1, W21p, 256, 64);

  const int eGrid   = (E * 32 + 255) / 256;
  const int nGrid   = ((int)(NF / 4) + 255) / 256;
  const int mlpGrid = (N + 31) / 32;

  // ---- layer 0 ----
  hipMemsetAsync(mI, 0, NF * 4 * 3, stream);  // zeros mF, sE, sME
  edge_max_kernel<<<eGrid, 256, 0, stream>>>(x, src, dstp, mI, E);
  edge_sum_kernel<<<eGrid, 256, 0, stream>>>(x, src, dstp, mF, sE, sME, E);
  node_finish_kernel<<<nGrid, 256, 0, stream>>>(x, sE, sME, hbf, (int)(NF / 4));
  mlp_kernel<128, true><<<mlpGrid, 256, 0, stream>>>(
      hbf, W10p, b1_0, g_0, be_0, W20p, b2_0, x1, N);

  // ---- layer 1 ----
  hipMemsetAsync(mI, 0, NF * 4 * 3, stream);
  edge_max_kernel<<<eGrid, 256, 0, stream>>>(x1, src, dstp, mI, E);
  edge_sum_kernel<<<eGrid, 256, 0, stream>>>(x1, src, dstp, mF, sE, sME, E);
  node_finish_kernel<<<nGrid, 256, 0, stream>>>(x1, sE, sME, hbf, (int)(NF / 4));
  mlp_kernel<64, false><<<mlpGrid, 256, 0, stream>>>(
      hbf, W11p, b1_1, g_1, be_1, W21p, b2_1, (float*)d_out, N);
}
